// LSTMModel_43825846289026
// MI455X (gfx1250) — matrix operand order repaired and checked by reading, not compiled, not run on hardware
//
#include <hip/hip_runtime.h>
#include <hip/hip_bf16.h>

typedef __attribute__((ext_vector_type(16))) __bf16 v16bf;
typedef __attribute__((ext_vector_type(8)))  float  v8f;

#define T_STEPS 512
#define UNITS   256
#define HIDDEN  100

__device__ __forceinline__ float sigm_f(float x) {
    return 1.0f / (1.0f + __expf(-x));
}
__device__ __forceinline__ float tanh_f(float x) {
    x = fminf(15.0f, fmaxf(-15.0f, x));
    float e = __expf(-2.0f * x);
    return (1.0f - e) / (1.0f + e);
}
__device__ __forceinline__ unsigned short f32_to_bf16(float f) {
    unsigned u = __builtin_bit_cast(unsigned, f);
    u += 0x7FFFu + ((u >> 16) & 1u);          // round-to-nearest-even
    return (unsigned short)(u >> 16);
}

// Pre-swizzle Wh [256,1024] f32 -> bf16 tiles in the contiguous-K WMMA B-operand
// register layout. Tile (nt, kc) covers cols [nt*16,nt*16+16), K [kc*32,kc*32+32).
// Within a tile: lane<16 -> N=lane,  K=elem;  lane>=16 -> N=lane-16, K=16+elem.
// Element (tile, lane, e) stored flat so each lane's v16bf is 32 contiguous bytes.
__global__ void prep_wh_kernel(const float* __restrict__ Wh,
                               unsigned short* __restrict__ whT) {
    int idx = blockIdx.x * 256 + threadIdx.x;
    if (idx >= 64 * 8 * 512) return;
    int tile = idx >> 9;
    int rem  = idx & 511;
    int lane = rem >> 4;
    int e    = rem & 15;
    int nt   = tile >> 3;
    int kc   = tile & 7;
    int k    = kc * 32 + e + 16 * (lane >> 4);
    int col  = nt * 16 + (lane & 15);
    whT[idx] = f32_to_bf16(Wh[k * (4 * UNITS) + col]);
}

// Persistent fused LSTM + dense head. One workgroup = 16 batch rows.
// 8 waves x (4 gates x 2 sub-tiles) = all 1024 gate columns, arranged so each
// wave owns units [32w, 32w+32) of every gate -> cell update is in-wave.
__global__ __launch_bounds__(256) void lstm_fused_kernel(
    const float* __restrict__ x,       // [1024, 512]
    const float* __restrict__ Wi,      // [1024]   (F == 1)
    const unsigned short* __restrict__ whT, // swizzled bf16 Wh
    const float* __restrict__ bias,    // [1024]
    const float* __restrict__ W1,      // [256,100]
    const float* __restrict__ b1,      // [100]
    const float* __restrict__ W2,      // [100]
    const float* __restrict__ b2,      // [1]
    float* __restrict__ out)           // [1024]
{
    // h tile in the contiguous-K WMMA A-operand layout: [kc][lane][elem] bf16.
    __shared__ alignas(64) unsigned short hA[8 * 32 * 16];
    __shared__ float xs[16];           // x_t broadcast for the 16 rows
    __shared__ float hF[16 * UNITS];   // final-step h in f32 for dense head
    __shared__ float xh[16 * HIDDEN];

    const int tid    = threadIdx.x;
    const int wave   = tid >> 5;
    const int lane   = tid & 31;
    const int laneHi = lane >> 4;      // selects M rows 8..15 / K-half
    const int laneLo = lane & 15;      // N column within tile
    const int b0     = blockIdx.x * 16;

    const v16bf* hAV = reinterpret_cast<const v16bf*>(hA);
    const v16bf* whV = reinterpret_cast<const v16bf*>(whT);

    // Hoist per-tile Wi / bias (loop-invariant over t).
    float wiR[4][2], bR[4][2];
    int ntl[4][2];
    #pragma unroll
    for (int g = 0; g < 4; ++g) {
        #pragma unroll
        for (int s = 0; s < 2; ++s) {
            int nt = g * 16 + wave * 2 + s;
            ntl[g][s] = nt;
            int col = nt * 16 + laneLo;
            wiR[g][s] = Wi[col];
            bR[g][s]  = bias[col];
        }
    }

    // h0 = 0, c0 = 0
    for (int i = tid; i < 8 * 32 * 16; i += 256) hA[i] = 0;
    if (tid < 16) xs[tid] = x[(b0 + tid) * T_STEPS];
    float creg[2][8];
    #pragma unroll
    for (int s = 0; s < 2; ++s)
        #pragma unroll
        for (int v = 0; v < 8; ++v) creg[s][v] = 0.0f;
    __syncthreads();

    #pragma unroll 1
    for (int t = 0; t < T_STEPS; ++t) {
        // Opaque zero offset: forces the Wh tile loads to be loop-variant so
        // LICM cannot hoist 512 VGPRs worth of B tiles out of the loop and
        // spill them to scratch. Runtime value is always 0; the same 512 KB
        // of Wh streams from WGP$/L2 every step instead.
        int zoff = t;
        asm volatile("" : "+v"(zoff));
        zoff -= t;
        const v16bf* whB = whV + zoff;

        // z init: x_t * Wi + b, laid out in C-operand register order
        // (lane L, VGPR v) -> row m = v + 8*(L/16), col n = L%16.
        float xr[8];
        #pragma unroll
        for (int v = 0; v < 8; ++v) xr[v] = xs[v + 8 * laneHi];

        v8f acc[4][2];
        #pragma unroll
        for (int g = 0; g < 4; ++g)
            #pragma unroll
            for (int s = 0; s < 2; ++s)
                #pragma unroll
                for (int v = 0; v < 8; ++v)
                    acc[g][s][v] = xr[v] * wiR[g][s] + bR[g][s];

        // z += h @ Wh : 8 K-chunks x 8 tiles of v_wmma_f32_16x16x32_bf16
        #pragma unroll
        for (int kc = 0; kc < 8; ++kc) {
            v16bf a = hAV[kc * 32 + lane];           // shared A across 8 tiles
            v16bf bm[4][2];
            #pragma unroll
            for (int g = 0; g < 4; ++g)
                #pragma unroll
                for (int s = 0; s < 2; ++s)
                    bm[g][s] = whB[(ntl[g][s] * 8 + kc) * 32 + lane];
            #pragma unroll
            for (int g = 0; g < 4; ++g)
                #pragma unroll
                for (int s = 0; s < 2; ++s)
                    acc[g][s] = __builtin_amdgcn_wmma_f32_16x16x32_bf16(
                        false, a, false, bm[g][s], (short)0, acc[g][s],
                        false, false);
        }
        __syncthreads();   // all reads of hA done

        // In-wave LSTM cell update: this wave owns units wave*32 .. wave*32+31.
        #pragma unroll
        for (int s = 0; s < 2; ++s) {
            const int u   = wave * 32 + s * 16 + laneLo;
            const int kc  = u >> 5;
            const int sub = u & 31;
            const int lh  = sub >> 4;
            const int e   = sub & 15;
            #pragma unroll
            for (int v = 0; v < 8; ++v) {
                const int brow = v + 8 * laneHi;
                float ig = sigm_f(acc[0][s][v]);
                float fg = sigm_f(acc[1][s][v]);
                float gg = tanh_f(acc[2][s][v]);
                float og = sigm_f(acc[3][s][v]);
                float cn = fg * creg[s][v] + ig * gg;
                creg[s][v] = cn;
                float hv = og * tanh_f(cn);
                // write h back into the A-operand layout for next step
                hA[(kc * 32 + brow + 16 * lh) * 16 + e] = f32_to_bf16(hv);
                if (t == T_STEPS - 1) hF[brow * UNITS + u] = hv;
            }
        }
        if (tid < 16 && t + 1 < T_STEPS)
            xs[tid] = x[(b0 + tid) * T_STEPS + t + 1];
        __syncthreads();   // h(t+1), x(t+1) visible
    }

    // Dense head: relu(h @ W1 + b1) then sigmoid(. @ W2 + b2). Tiny -> VALU.
    for (int idx = tid; idx < 16 * HIDDEN; idx += 256) {
        int brow = idx / HIDDEN;
        int hh   = idx % HIDDEN;
        float s = b1[hh];
        #pragma unroll 4
        for (int k = 0; k < UNITS; ++k)
            s += hF[brow * UNITS + k] * W1[k * HIDDEN + hh];
        xh[idx] = fmaxf(s, 0.0f);
    }
    __syncthreads();
    if (tid < 16) {
        float s = b2[0];
        #pragma unroll 4
        for (int hh = 0; hh < HIDDEN; ++hh)
            s += xh[tid * HIDDEN + hh] * W2[hh];
        out[b0 + tid] = sigm_f(s);
    }
}

extern "C" void kernel_launch(void* const* d_in, const int* in_sizes, int n_in,
                              void* d_out, int out_size, void* d_ws, size_t ws_size,
                              hipStream_t stream) {
    const float* x    = (const float*)d_in[0];
    const float* Wi   = (const float*)d_in[1];
    const float* Wh   = (const float*)d_in[2];
    const float* bias = (const float*)d_in[3];
    const float* W1   = (const float*)d_in[4];
    const float* b1   = (const float*)d_in[5];
    const float* W2   = (const float*)d_in[6];
    const float* b2   = (const float*)d_in[7];
    float* out = (float*)d_out;
    unsigned short* whT = (unsigned short*)d_ws;   // 64*8*512 bf16 = 512 KB

    prep_wh_kernel<<<1024, 256, 0, stream>>>(Wh, whT);
    lstm_fused_kernel<<<64, 256, 0, stream>>>(x, Wi, whT, bias, W1, b1, W2, b2, out);
}
